// NodeSchNetBackbone_43963285242306
// MI455X (gfx1250) — matrix operand order repaired and checked by reading, not compiled, not run on hardware
//
#include <hip/hip_runtime.h>

// ---------------------------------------------------------------------------
// SchNet backbone on MI455X (gfx1250): all GEMMs via v_wmma_f32_16x16x32_bf16
// 128x64 block tile, 8 waves (32x32 each), double-buffered LDS pipeline.
// ---------------------------------------------------------------------------

typedef __bf16 bf16;
typedef __attribute__((ext_vector_type(16))) __bf16 v16bf;
typedef __attribute__((ext_vector_type(8)))  __bf16 v8bf;
typedef __attribute__((ext_vector_type(8)))  float  v8f;

#define NN   10000     // nodes
#define NE   100000    // edges
#define HID  600       // hidden channels
#define HP   640       // hidden padded (20 * 32)
#define NG   50        // gaussians
#define NGP  64        // gaussians padded (2 * 32)
#define NI   6         // interactions

#define PI_F 3.14159265358979f
#define LN2_F 0.69314718055994f

__device__ __forceinline__ float ssp(float x) {
    // shifted softplus: softplus(x) - ln(2), numerically stable
    float sp = (x > 20.0f) ? x : log1pf(__expf(x));
    return sp - LN2_F;
}

// ---------------------------------------------------------------------------
// Edge geometry: distance + cosine cutoff
// ---------------------------------------------------------------------------
__global__ void edge_dist_kernel(const float* __restrict__ pos,
                                 const int* __restrict__ ei,
                                 float* __restrict__ dist,
                                 float* __restrict__ cvals) {
    int e = blockIdx.x * blockDim.x + threadIdx.x;
    if (e >= NE) return;
    int r = ei[e];
    int c = ei[NE + e];
    float dx = pos[r * 3 + 0] - pos[c * 3 + 0];
    float dy = pos[r * 3 + 1] - pos[c * 3 + 1];
    float dz = pos[r * 3 + 2] - pos[c * 3 + 2];
    float d = sqrtf(dx * dx + dy * dy + dz * dz);
    dist[e] = d;
    cvals[e] = 0.5f * (__cosf(d * PI_F / 10.0f) + 1.0f);
}

// ---------------------------------------------------------------------------
// Gaussian smearing -> bf16, zero padded to NGP columns
// ---------------------------------------------------------------------------
__global__ void edge_attr_kernel(const float* __restrict__ dist,
                                 bf16* __restrict__ ea) {
    int idx = blockIdx.x * blockDim.x + threadIdx.x;
    if (idx >= NE * NGP) return;
    int e = idx >> 6;
    int c = idx & 63;
    float v = 0.0f;
    if (c < NG) {
        const float step  = 10.0f / 49.0f;
        const float coeff = -0.5f / (step * step);
        float t = dist[e] - step * (float)c;
        v = __expf(coeff * t * t);
    }
    ea[idx] = (bf16)v;
}

// ---------------------------------------------------------------------------
// Embedding gather: fp32 master h (d_out) + bf16 padded copy
// ---------------------------------------------------------------------------
__global__ void embed_kernel(const int* __restrict__ z,
                             const float* __restrict__ emb,
                             float* __restrict__ hf,
                             bf16* __restrict__ hb) {
    int n = blockIdx.x;
    int zr = z[n] * HID;
    for (int c = threadIdx.x; c < HP; c += blockDim.x) {
        if (c < HID) {
            float v = emb[zr + c];
            hf[(size_t)n * HID + c] = v;
            hb[(size_t)n * HP + c] = (bf16)v;
        } else {
            hb[(size_t)n * HP + c] = (bf16)0.0f;
        }
    }
}

// ---------------------------------------------------------------------------
// Weight pad + fp32 -> bf16:  W [Ovalid, Kvalid] -> Wp [gridDim rows, Kp]
// ---------------------------------------------------------------------------
__global__ void pad_w_kernel(const float* __restrict__ w,
                             bf16* __restrict__ wp,
                             int Ovalid, int Kvalid, int Kp) {
    int o = blockIdx.x;   // 0..Np-1 (padded output rows)
    for (int k = threadIdx.x; k < Kp; k += blockDim.x) {
        float v = (o < Ovalid && k < Kvalid) ? w[(size_t)o * Kvalid + k] : 0.0f;
        wp[(size_t)o * Kp + k] = (bf16)v;
    }
}

// ---------------------------------------------------------------------------
// Zero fp32 buffer
// ---------------------------------------------------------------------------
__global__ void zero_kernel(float* __restrict__ p, int count) {
    int i = blockIdx.x * blockDim.x + threadIdx.x;
    if (i < count) p[i] = 0.0f;
}

// ---------------------------------------------------------------------------
// fp32 [N,HID] -> bf16 padded [N,HP]
// ---------------------------------------------------------------------------
__global__ void cvt_pad_kernel(const float* __restrict__ src,
                               bf16* __restrict__ dst) {
    int n = blockIdx.x;
    for (int c = threadIdx.x; c < HP; c += blockDim.x) {
        float v = (c < HID) ? src[(size_t)n * HID + c] : 0.0f;
        dst[(size_t)n * HP + c] = (bf16)v;
    }
}

// ---------------------------------------------------------------------------
// Message passing: m = x[row] * W  scattered-add into agg[col]
// ---------------------------------------------------------------------------
__global__ void scatter_kernel(const int* __restrict__ ei,
                               const bf16* __restrict__ xb,
                               const bf16* __restrict__ W,
                               float* __restrict__ agg) {
    int e = blockIdx.x;
    int r = ei[e];
    int c = ei[NE + e];
    const bf16* xr = xb + (size_t)r * HP;
    const bf16* we = W + (size_t)e * HP;
    float* ag = agg + (size_t)c * HID;
    for (int j = threadIdx.x; j < HID; j += blockDim.x) {
        atomicAdd(&ag[j], (float)xr[j] * (float)we[j]);
    }
}

// ---------------------------------------------------------------------------
// WMMA GEMM:  out = epilogue( A[M,Kp] @ B[Np,Kp]^T + bias )
//   A row-major bf16 (lda == Kp), B = padded weight matrix (row o = output
//   channel o's input weights). Block tile 128x64, 8 waves (4x2), each wave
//   computes a 32x32 subtile = 4 accumulators / 4 WMMAs per K-step.
//   Double-buffered LDS, one barrier per K-step.
// ---------------------------------------------------------------------------
#define BM 128
#define BN 64
#define BK 32

#define EPI_PLAIN 0   // out_bf = acc (pad cols -> 0)
#define EPI_SSP   1   // out_bf = ssp(acc + bias)
#define EPI_SCALE 2   // out_bf = (acc + bias) * scale[row]
#define EPI_RESID 3   // out_f += acc + bias ; out_bf = out_f   (residual h)

__global__ __launch_bounds__(256)
void wmma_gemm_kernel(const bf16* __restrict__ A,
                      const bf16* __restrict__ B,
                      const float* __restrict__ bias,
                      const float* __restrict__ scale,
                      bf16* __restrict__ outb,
                      float* __restrict__ outf,
                      int M, int Kp, int Nvalid, int mode) {
    __shared__ bf16 lA[2][BM * BK];   // 2 x 8 KB
    __shared__ bf16 lB[2][BN * BK];   // 2 x 4 KB

    const int tid  = threadIdx.x;
    const int lane = tid & 31;
    const int wave = tid >> 5;
    const int wm   = wave & 3;   // wave row 0..3  (32 rows each)
    const int wn   = wave >> 2;  // wave col 0..1  (32 cols each)
    const int m0   = blockIdx.x * BM;
    const int n0   = blockIdx.y * BN;

    // cooperative global->LDS staging indices
    const int lrA = tid >> 1;          // 0..127
    const int lcA = (tid & 1) * 16;    // 0 or 16  (16 bf16 per thread)
    const int lrB = tid >> 2;          // 0..63
    const int lcB = (tid & 3) * 8;     // 8 bf16 per thread

    const bool aValid = (m0 + lrA) < M;
    const bf16* gA = A + (size_t)(m0 + lrA) * Kp + lcA;
    const bf16* gB = B + (size_t)(n0 + lrB) * Kp + lcB;   // weights fully padded

    v8f acc[2][2] = {};

    // ---- preload K-step 0 ----
    v8bf rA0 = {}, rA1 = {}, rB = {};
    if (aValid) {
        rA0 = *(const v8bf*)(gA);
        rA1 = *(const v8bf*)(gA + 8);
    }
    rB = *(const v8bf*)(gB);
    *(v8bf*)(&lA[0][lrA * BK + lcA])     = rA0;
    *(v8bf*)(&lA[0][lrA * BK + lcA + 8]) = rA1;
    *(v8bf*)(&lB[0][lrB * BK + lcB])     = rB;
    __syncthreads();

    const int nsteps = Kp / BK;   // Kp is always a multiple of 32
    for (int s = 0; s < nsteps; ++s) {
        const int buf = s & 1;

        // ---- stage next K-step into registers (overlaps with WMMA) ----
        if (s + 1 < nsteps) {
            const int k = (s + 1) * BK;
            rA0 = (v8bf){}; rA1 = (v8bf){};
            if (aValid) {
                rA0 = *(const v8bf*)(gA + k);
                rA1 = *(const v8bf*)(gA + k + 8);
                if (s + 2 < nsteps)
                    __builtin_prefetch(gA + k + BK, 0, 0);
            }
            rB = *(const v8bf*)(gB + k);
        }

        // ---- build fragments from current LDS buffer ----
        const bf16* lab = &lA[buf][0];
        const bf16* lbb = &lB[buf][0];
        const int ar = lane & 15;
        const int ak = (lane < 16) ? 0 : 8;
        const int bk = (lane < 16) ? 0 : 8;

        v16bf af[2], bg[2];
#pragma unroll
        for (int t = 0; t < 2; ++t) {
            const bf16* ap = lab + (wm * 32 + t * 16 + ar) * BK + ak;
            v8bf x0 = *(const v8bf*)(ap);
            v8bf x1 = *(const v8bf*)(ap + 16);
            af[t] = __builtin_shufflevector(x0, x1,
                0, 1, 2, 3, 4, 5, 6, 7, 8, 9, 10, 11, 12, 13, 14, 15);
            const bf16* bp = lbb + (wn * 32 + t * 16 + ar) * BK + bk;
            v8bf y0 = *(const v8bf*)(bp);
            v8bf y1 = *(const v8bf*)(bp + 16);
            bg[t] = __builtin_shufflevector(y0, y1,
                0, 1, 2, 3, 4, 5, 6, 7, 8, 9, 10, 11, 12, 13, 14, 15);
        }

        // ---- 4 independent WMMAs (covers bf16 WMMA hazard slots) ----
        acc[0][0] = __builtin_amdgcn_wmma_f32_16x16x32_bf16(
            false, af[0], false, bg[0], (short)0, acc[0][0], false, false);
        acc[0][1] = __builtin_amdgcn_wmma_f32_16x16x32_bf16(
            false, af[0], false, bg[1], (short)0, acc[0][1], false, false);
        acc[1][0] = __builtin_amdgcn_wmma_f32_16x16x32_bf16(
            false, af[1], false, bg[0], (short)0, acc[1][0], false, false);
        acc[1][1] = __builtin_amdgcn_wmma_f32_16x16x32_bf16(
            false, af[1], false, bg[1], (short)0, acc[1][1], false, false);

        // ---- write next tile into the other buffer ----
        if (s + 1 < nsteps) {
            const int nbuf = buf ^ 1;
            *(v8bf*)(&lA[nbuf][lrA * BK + lcA])     = rA0;
            *(v8bf*)(&lA[nbuf][lrA * BK + lcA + 8]) = rA1;
            *(v8bf*)(&lB[nbuf][lrB * BK + lcB])     = rB;
        }
        __syncthreads();
    }

    // ---- epilogue: D layout -> lane (N = lane%16), vgpr v (M = v + 8*(lane>=16))
    const int rb0 = m0 + wm * 32 + ((lane < 16) ? 0 : 8);
    const int cb0 = n0 + wn * 32 + (lane & 15);
#pragma unroll
    for (int t = 0; t < 2; ++t) {
#pragma unroll
        for (int u = 0; u < 2; ++u) {
            const int col = cb0 + u * 16;
            const bool cvalid = (col < Nvalid);
            const float bv = (cvalid && bias) ? bias[col] : 0.0f;
#pragma unroll
            for (int v = 0; v < 8; ++v) {
                const int row = rb0 + t * 16 + v;
                if (row >= M) continue;
                if (!cvalid) {
                    outb[(size_t)row * HP + col] = (bf16)0.0f;
                    continue;
                }
                float val = acc[t][u][v] + bv;
                if (mode == EPI_SSP) {
                    val = ssp(val);
                } else if (mode == EPI_SCALE) {
                    val *= scale[row];
                } else if (mode == EPI_RESID) {
                    val += outf[(size_t)row * Nvalid + col];
                    outf[(size_t)row * Nvalid + col] = val;
                }
                outb[(size_t)row * HP + col] = (bf16)val;
            }
        }
    }
}

// ---------------------------------------------------------------------------
// Host-side orchestration
// ---------------------------------------------------------------------------
extern "C" void kernel_launch(void* const* d_in, const int* in_sizes, int n_in,
                              void* d_out, int out_size, void* d_ws, size_t ws_size,
                              hipStream_t stream) {
    const int*   z      = (const int*)d_in[0];
    const float* pos    = (const float*)d_in[1];
    const int*   ei     = (const int*)d_in[2];
    const float* emb    = (const float*)d_in[3];
    const float* mlp1_w = (const float*)d_in[4];
    const float* mlp1_b = (const float*)d_in[5];
    const float* mlp2_w = (const float*)d_in[6];
    const float* mlp2_b = (const float*)d_in[7];
    const float* lin1_w = (const float*)d_in[8];
    const float* lin2_w = (const float*)d_in[9];
    const float* lin2_b = (const float*)d_in[10];
    const float* lin_w  = (const float*)d_in[11];
    const float* lin_b  = (const float*)d_in[12];
    float* hf = (float*)d_out;   // fp32 master h, [NN, HID]

    // ---- workspace carve-out ----
    char* p = (char*)d_ws;
    auto alloc = [&](size_t bytes) -> char* {
        char* r = p;
        p += (bytes + 255) & ~(size_t)255;
        return r;
    };
    float* dist  = (float*)alloc((size_t)NE * 4);
    float* cvals = (float*)alloc((size_t)NE * 4);
    bf16*  ea    = (bf16*) alloc((size_t)NE * NGP * 2);
    bf16*  hb    = (bf16*) alloc((size_t)NN * HP * 2);
    bf16*  W1    = (bf16*) alloc((size_t)NE * HP * 2);
    bf16*  Wf    = (bf16*) alloc((size_t)NE * HP * 2);
    bf16*  xb    = (bf16*) alloc((size_t)NN * HP * 2);
    float* agg   = (float*)alloc((size_t)NN * HID * 4);
    bf16*  aggb  = (bf16*) alloc((size_t)NN * HP * 2);
    bf16*  x3    = (bf16*) alloc((size_t)NN * HP * 2);
    bf16*  B1p   = (bf16*) alloc((size_t)HP * NGP * 2);
    bf16*  B2p   = (bf16*) alloc((size_t)HP * HP * 2);
    bf16*  L1p   = (bf16*) alloc((size_t)HP * HP * 2);
    bf16*  L2p   = (bf16*) alloc((size_t)HP * HP * 2);
    bf16*  Lp    = (bf16*) alloc((size_t)HP * HP * 2);

    // ---- edge geometry (once) ----
    edge_dist_kernel<<<(NE + 255) / 256, 256, 0, stream>>>(pos, ei, dist, cvals);
    edge_attr_kernel<<<(NE * NGP) / 256, 256, 0, stream>>>(dist, ea);
    embed_kernel<<<NN, 256, 0, stream>>>(z, emb, hf, hb);

    const dim3 gEdge((NE + BM - 1) / BM, HP / BN);
    const dim3 gNode((NN + BM - 1) / BM, HP / BN);

    for (int i = 0; i < NI; ++i) {
        // pad/convert this interaction's weights
        pad_w_kernel<<<HP, 256, 0, stream>>>(mlp1_w + (size_t)i * HID * NG, B1p, HID, NG, NGP);
        pad_w_kernel<<<HP, 256, 0, stream>>>(mlp2_w + (size_t)i * HID * HID, B2p, HID, HID, HP);
        pad_w_kernel<<<HP, 256, 0, stream>>>(lin1_w + (size_t)i * HID * HID, L1p, HID, HID, HP);
        pad_w_kernel<<<HP, 256, 0, stream>>>(lin2_w + (size_t)i * HID * HID, L2p, HID, HID, HP);
        pad_w_kernel<<<HP, 256, 0, stream>>>(lin_w  + (size_t)i * HID * HID, Lp,  HID, HID, HP);

        // W1 = ssp(edge_attr @ mlp1^T + b1)        [E, HP]
        wmma_gemm_kernel<<<gEdge, 256, 0, stream>>>(
            ea, B1p, mlp1_b + (size_t)i * HID, nullptr, W1, nullptr,
            NE, NGP, HID, EPI_SSP);
        // W = (W1 @ mlp2^T + b2) * C[edge]         [E, HP]
        wmma_gemm_kernel<<<gEdge, 256, 0, stream>>>(
            W1, B2p, mlp2_b + (size_t)i * HID, cvals, Wf, nullptr,
            NE, HP, HID, EPI_SCALE);
        // x = h @ lin1^T (no bias)                 [N, HP]
        wmma_gemm_kernel<<<gNode, 256, 0, stream>>>(
            hb, L1p, nullptr, nullptr, xb, nullptr,
            NN, HP, HID, EPI_PLAIN);

        // agg[col] += x[row] * W
        zero_kernel<<<((size_t)NN * HID + 255) / 256, 256, 0, stream>>>(agg, NN * HID);
        scatter_kernel<<<NE, 256, 0, stream>>>(ei, xb, Wf, agg);
        cvt_pad_kernel<<<NN, 256, 0, stream>>>(agg, aggb);

        // x3 = ssp(agg @ lin2^T + b)               [N, HP]
        wmma_gemm_kernel<<<gNode, 256, 0, stream>>>(
            aggb, L2p, lin2_b + (size_t)i * HID, nullptr, x3, nullptr,
            NN, HP, HID, EPI_SSP);
        // h += x3 @ lin^T + b  (fp32 master + bf16 copy)
        wmma_gemm_kernel<<<gNode, 256, 0, stream>>>(
            x3, Lp, lin_b + (size_t)i * HID, nullptr, hb, hf,
            NN, HP, HID, EPI_RESID);
    }
}
